// Quant_Linear_Int_43877385896252
// MI455X (gfx1250) — compile-verified, matrix operand order checked by reading, not run on hardware
//
#include <hip/hip_runtime.h>

#define TOKENS 8192
#define IN_F   4096
#define OUT_F  4096
#define KD     IN_F
#define EPSQ   1e-8f

typedef int v8i __attribute__((ext_vector_type(8)));

// ---------------- workspace layout (bytes) ----------------
#define QX_OFF   ((size_t)0)                                  // 8192*4096 int8  = 32 MB
#define QW_OFF   (QX_OFF + (size_t)TOKENS * IN_F)             // 4096*4096 int8  = 16 MB
#define SW_OFF   (QW_OFF + (size_t)OUT_F * IN_F)              // 4096 f32
#define ZW_OFF   (SW_OFF + (size_t)OUT_F * 4)                 // 4096 f32
#define WSUM_OFF (ZW_OFF + (size_t)OUT_F * 4)                 // 4096 f32
#define XSUM_OFF (WSUM_OFF + (size_t)OUT_F * 4)               // 8192 f32

__device__ __forceinline__ int clampq(int q, int h) {
    return q < -h ? -h : (q > h - 1 ? h - 1 : q);
}

// ---------------- pass 1: per-row weight quantization ----------------
__global__ __launch_bounds__(256)
void quant_weight_kernel(const float* __restrict__ w,
                         unsigned int* __restrict__ qw,
                         float* __restrict__ sw, float* __restrict__ zw,
                         float* __restrict__ wsum) {
    const int row = blockIdx.x;
    const int tid = threadIdx.x;
    const float4* wr = (const float4*)(w + (size_t)row * IN_F);

    float4 v[4];
    float lmin = 3.4e38f, lmax = -3.4e38f;
#pragma unroll
    for (int j = 0; j < 4; ++j) {
        v[j] = wr[tid + j * 256];
        lmin = fminf(lmin, fminf(fminf(v[j].x, v[j].y), fminf(v[j].z, v[j].w)));
        lmax = fmaxf(lmax, fmaxf(fmaxf(v[j].x, v[j].y), fmaxf(v[j].z, v[j].w)));
    }

    __shared__ float red[256];
    __shared__ int   redi[256];
    red[tid] = lmin; __syncthreads();
    for (int s = 128; s > 0; s >>= 1) {
        if (tid < s) red[tid] = fminf(red[tid], red[tid + s]);
        __syncthreads();
    }
    const float rmin = red[0]; __syncthreads();
    red[tid] = lmax; __syncthreads();
    for (int s = 128; s > 0; s >>= 1) {
        if (tid < s) red[tid] = fmaxf(red[tid], red[tid + s]);
        __syncthreads();
    }
    const float rmax = red[0]; __syncthreads();

    const float scale = 255.0f / fmaxf(rmax - rmin, EPSQ);
    const float zp    = rintf(scale * rmin) + 128.0f;

    unsigned int* qr = qw + (size_t)row * (IN_F / 4);
    int ssum = 0;
#pragma unroll
    for (int j = 0; j < 4; ++j) {
        const int d = tid + j * 256;
        int q0 = clampq((int)rintf(scale * v[j].x - zp), 128);
        int q1 = clampq((int)rintf(scale * v[j].y - zp), 128);
        int q2 = clampq((int)rintf(scale * v[j].z - zp), 128);
        int q3 = clampq((int)rintf(scale * v[j].w - zp), 128);
        ssum += q0 + q1 + q2 + q3;
        qr[d] = (unsigned)(q0 & 0xff) | ((unsigned)(q1 & 0xff) << 8) |
                ((unsigned)(q2 & 0xff) << 16) | ((unsigned)(q3 & 0xff) << 24);
    }
    redi[tid] = ssum; __syncthreads();
    for (int s = 128; s > 0; s >>= 1) {
        if (tid < s) redi[tid] += redi[tid + s];
        __syncthreads();
    }
    if (tid == 0) {
        sw[row]   = scale;
        zw[row]   = zp;
        wsum[row] = (float)redi[0];
    }
}

// ---------------- pass 2: per-tensor activation quantization ----------------
__global__ __launch_bounds__(256)
void quant_x_kernel(const float* __restrict__ x,
                    unsigned int* __restrict__ qx,
                    float* __restrict__ xsum,
                    const float* __restrict__ xmin_p,
                    const float* __restrict__ xmax_p,
                    const int* __restrict__ abit_p) {
    const int row = blockIdx.x;
    const int tid = threadIdx.x;
    const int ab  = *abit_p;
    const float nlev = (float)((1 << ab) - 1);
    const int   halfq = 1 << (ab - 1);
    const float xmn = *xmin_p, xmx = *xmax_p;
    const float s = nlev / fmaxf(xmx - xmn, EPSQ);
    const float z = rintf(s * xmn) + (float)halfq;

    const float4* xr = (const float4*)(x + (size_t)row * IN_F);
    unsigned int* qr = qx + (size_t)row * (IN_F / 4);

    int ssum = 0;
#pragma unroll
    for (int j = 0; j < 4; ++j) {
        const int d = tid + j * 256;
        float4 v = xr[d];
        int q0 = clampq((int)rintf(s * v.x - z), halfq);
        int q1 = clampq((int)rintf(s * v.y - z), halfq);
        int q2 = clampq((int)rintf(s * v.z - z), halfq);
        int q3 = clampq((int)rintf(s * v.w - z), halfq);
        ssum += q0 + q1 + q2 + q3;
        qr[d] = (unsigned)(q0 & 0xff) | ((unsigned)(q1 & 0xff) << 8) |
                ((unsigned)(q2 & 0xff) << 16) | ((unsigned)(q3 & 0xff) << 24);
    }
    __shared__ int redi[256];
    redi[tid] = ssum; __syncthreads();
    for (int st = 128; st > 0; st >>= 1) {
        if (tid < st) redi[tid] += redi[tid + st];
        __syncthreads();
    }
    if (tid == 0) xsum[row] = (float)redi[0];
}

// ---------------- pass 3: int8 WMMA GEMM + dequant epilogue ----------------
// LDS tile rows padded to 80B: bank-conflict-free for the WMMA fragment
// ds_load_b64 pattern, and 16B-aligned (80 = 5*16) for async b128 fills.
#define LROW  80
#define BUFSZ (128 * LROW)

// CDNA5 async copy: global -> LDS, no VGPR round trip, tracked by ASYNCcnt.
// GVS addressing: mem = SGPR64 + VGPR32 + ioffset.
__device__ __forceinline__ void async_copy16(unsigned lds_off,
                                             const unsigned char* gbase,
                                             unsigned goff) {
    asm volatile("global_load_async_to_lds_b128 %0, %1, %2 offset:0"
                 :
                 : "v"(lds_off), "v"(goff), "s"(gbase)
                 : "memory");
}
__device__ __forceinline__ void wait_async0() {
    asm volatile("s_wait_asynccnt 0" ::: "memory");
}

__device__ __forceinline__ v8i load_frag(const unsigned char* p) {
    // p = row base + 8*(lane>=16). ISA 8-bit A/B layout: dword-pairs at byte
    // offsets {0,16,32,48} (+8 for the upper half-wave) of a 64B K-row.
    const unsigned long long* q = (const unsigned long long*)p;
    union { v8i v; unsigned long long u[4]; } f;
    f.u[0] = q[0];
    f.u[1] = q[2];
    f.u[2] = q[4];
    f.u[3] = q[6];
    return f.v;
}

__global__ __launch_bounds__(256)
void gemm_i8_kernel(const unsigned char* __restrict__ qx,
                    const unsigned char* __restrict__ qw,
                    const float* __restrict__ sw, const float* __restrict__ zw,
                    const float* __restrict__ wsum, const float* __restrict__ xsum,
                    const float* __restrict__ bias,
                    const float* __restrict__ xmin_p, const float* __restrict__ xmax_p,
                    const int* __restrict__ abit_p,
                    float* __restrict__ out) {
    // double-buffered tiles: 2 x (128 rows x 64 K-bytes, padded) for A and B
    __shared__ unsigned char ldsA[2 * BUFSZ];
    __shared__ unsigned char ldsB[2 * BUFSZ];

    const int tid   = threadIdx.x;
    const int ntile = blockIdx.x;                // N/128 = 32
    const int mtile = blockIdx.y;                // M/128 = 64
    const int wave  = tid >> 5;
    const int lane  = tid & 31;
    const int wm    = wave & 3;                  // wave M position (0..3) -> 32 rows
    const int wn    = wave >> 2;                 // wave N position (0..1) -> 64 cols
    const int lrow  = lane & 15;
    const int lhalf = lane >> 4;

    v8i acc[2][4];
#pragma unroll
    for (int a = 0; a < 2; ++a)
#pragma unroll
        for (int b = 0; b < 4; ++b)
#pragma unroll
            for (int i = 0; i < 8; ++i) acc[a][b][i] = 0;

    const unsigned char* gA = qx + (size_t)mtile * 128 * KD;
    const unsigned char* gB = qw + (size_t)ntile * 128 * KD;
    const int r0 = tid >> 2;                     // 0..63
    const int c0 = tid & 3;                      // 16B chunk within 64B K-row

    // AMDGPU generic LDS pointer = aperture_hi:offset32 -> low 32 bits = LDS addr
    const unsigned ldsA_base = (unsigned)(size_t)(void*)ldsA;
    const unsigned ldsB_base = (unsigned)(size_t)(void*)ldsB;
    const unsigned loff0 = (unsigned)(r0 * LROW + c0 * 16);
    const unsigned loff1 = (unsigned)((r0 + 64) * LROW + c0 * 16);
    const unsigned goff0 = (unsigned)(r0 * KD + c0 * 16);
    const unsigned goff1 = (unsigned)((r0 + 64) * KD + c0 * 16);

    // prologue: fill buffer 0 for kt = 0
    async_copy16(ldsA_base + loff0, gA, goff0);
    async_copy16(ldsA_base + loff1, gA, goff1);
    async_copy16(ldsB_base + loff0, gB, goff0);
    async_copy16(ldsB_base + loff1, gB, goff1);
    wait_async0();
    __syncthreads();

    int cur = 0;
#pragma unroll 1
    for (int kt = 0; kt < KD; kt += 64) {
        // kick off async fill of the other buffer for the next K-chunk
        if (kt + 64 < KD) {
            const unsigned nb = (unsigned)((cur ^ 1) * BUFSZ);
            const unsigned gk = (unsigned)(kt + 64);
            async_copy16(ldsA_base + nb + loff0, gA, goff0 + gk);
            async_copy16(ldsA_base + nb + loff1, gA, goff1 + gk);
            async_copy16(ldsB_base + nb + loff0, gB, goff0 + gk);
            async_copy16(ldsB_base + nb + loff1, gB, goff1 + gk);
        }

        // compute on the current buffer (overlaps with the async fill)
        const unsigned char* bufA = ldsA + cur * BUFSZ;
        const unsigned char* bufB = ldsB + cur * BUFSZ;
        v8i afrag[2];
#pragma unroll
        for (int mi = 0; mi < 2; ++mi)
            afrag[mi] = load_frag(bufA + (wm * 32 + mi * 16 + lrow) * LROW + lhalf * 8);

#pragma unroll
        for (int ni = 0; ni < 4; ++ni) {
            v8i bfrag = load_frag(bufB + (wn * 64 + ni * 16 + lrow) * LROW + lhalf * 8);
            acc[0][ni] = __builtin_amdgcn_wmma_i32_16x16x64_iu8(
                true, afrag[0], true, bfrag, acc[0][ni], false, false);
            acc[1][ni] = __builtin_amdgcn_wmma_i32_16x16x64_iu8(
                true, afrag[1], true, bfrag, acc[1][ni], false, false);
        }

        // single barrier per K-step: async fill done + all waves done reading
        wait_async0();
        __syncthreads();
        cur ^= 1;
    }

    // dequant epilogue: out = (acc + z_w*sum_qx + z_x*sum_qw + z_x*z_w*K)/(s_x*s_w) + bias
    const int ab = abit_p[0];
    const float nlev = (float)((1 << ab) - 1);
    const float xmn = xmin_p[0], xmx = xmax_p[0];
    const float s_x = nlev / fmaxf(xmx - xmn, EPSQ);
    const float z_x = rintf(s_x * xmn) + (float)(1 << (ab - 1));

#pragma unroll
    for (int ni = 0; ni < 4; ++ni) {
        const int n = ntile * 128 + wn * 64 + ni * 16 + lrow;
        const float swv = sw[n], zwv = zw[n];
        const float rs  = 1.0f / (s_x * swv);
        const float cz  = z_x * wsum[n] + z_x * zwv * (float)KD;
        const float bv  = bias[n];
#pragma unroll
        for (int mi = 0; mi < 2; ++mi) {
            const int mbase = mtile * 128 + wm * 32 + mi * 16 + lhalf * 8;
#pragma unroll
            for (int i = 0; i < 8; ++i) {
                const int m = mbase + i;  // C/D layout: vgpr i -> M=i (+8 upper half)
                const float corr = zwv * xsum[m] + cz;
                out[(size_t)m * OUT_F + n] = ((float)acc[mi][ni][i] + corr) * rs + bv;
            }
        }
    }
}

// ---------------- host-side launch ----------------
extern "C" void kernel_launch(void* const* d_in, const int* in_sizes, int n_in,
                              void* d_out, int out_size, void* d_ws, size_t ws_size,
                              hipStream_t stream) {
    const float* x      = (const float*)d_in[0];
    const float* weight = (const float*)d_in[1];
    const float* bias   = (const float*)d_in[2];
    const float* xmin   = (const float*)d_in[3];
    const float* xmax   = (const float*)d_in[4];
    const int*   abit   = (const int*)d_in[5];
    float* out = (float*)d_out;

    char* ws = (char*)d_ws;
    unsigned int* qX   = (unsigned int*)(ws + QX_OFF);
    unsigned int* qW   = (unsigned int*)(ws + QW_OFF);
    float*        sW   = (float*)(ws + SW_OFF);
    float*        zW   = (float*)(ws + ZW_OFF);
    float*        wSum = (float*)(ws + WSUM_OFF);
    float*        xSum = (float*)(ws + XSUM_OFF);

    quant_x_kernel<<<TOKENS, 256, 0, stream>>>(x, qX, xSum, xmin, xmax, abit);
    quant_weight_kernel<<<OUT_F, 256, 0, stream>>>(weight, qW, sW, zW, wSum);

    dim3 grid(OUT_F / 128, TOKENS / 128);
    gemm_i8_kernel<<<grid, 256, 0, stream>>>(
        (const unsigned char*)qX, (const unsigned char*)qW,
        sW, zW, wSum, xSum, bias, xmin, xmax, abit, out);
}
